// nestEGCNs_85263690760753
// MI455X (gfx1250) — matrix operand order repaired and checked by reading, not compiled, not run on hardware
//
#include <hip/hip_runtime.h>
#include <hip/hip_bf16.h>

typedef __attribute__((ext_vector_type(16))) _Float16 v16h;
typedef __attribute__((ext_vector_type(8)))  _Float16 v8h;
typedef __attribute__((ext_vector_type(4)))  _Float16 v4h;
typedef __attribute__((ext_vector_type(8)))  float    v8f;
typedef __attribute__((ext_vector_type(4)))  float    v4f;

#define N_NODES 4096
#define D_DIM   64
#define L_LAYERS 8
#define ADIM_   512
#define HEADS_  4
#define DH_     128
#define E_EDGES 131072
#define B_GRAPH 32
#define NPG_    128
#define KPOOL   16
#define EFG_    256

// ---------- helpers ----------
__device__ __forceinline__ unsigned f2ord(float f) {
  unsigned u = __float_as_uint(f);
  return (u >> 31) ? ~u : (u | 0x80000000u);
}
__device__ __forceinline__ float ord2f(unsigned u) {
  return (u >> 31) ? __uint_as_float(u & 0x7fffffffu) : __uint_as_float(~u);
}
__device__ __forceinline__ float gelu_f(float x) {
  return 0.5f * x * (1.0f + tanhf(0.7978845608f * (x + 0.044715f * x * x * x)));
}
__device__ __forceinline__ v4h cvt4(v4f x) {
  v4h r;
  r[0] = (_Float16)x[0]; r[1] = (_Float16)x[1];
  r[2] = (_Float16)x[2]; r[3] = (_Float16)x[3];
  return r;
}

// CDNA5 wave32 WMMA operand loaders (contiguous k order: element s of lane half h holds k = 16h + s)
// A: 16x32 f16 tile, row-major in LDS, ld halfs per row (ld % 8 == 0, base 16B aligned)
// per-lane data is two contiguous 16B runs -> ds_load_b128 x2
__device__ __forceinline__ v16h load_A16(const _Float16* s, int ld) {
  int lane = threadIdx.x & 31;
  int m = lane & 15;
  int koff = (lane >> 4) << 4;
  v8h lo = *(const v8h*)(s + m * ld + koff);
  v8h hi = *(const v8h*)(s + m * ld + koff + 8);
  return __builtin_shufflevector(lo, hi, 0, 1, 2, 3, 4, 5, 6, 7,
                                 8, 9, 10, 11, 12, 13, 14, 15);
}
// B: 32x16 f16 tile stored TRANSPOSED in LDS as Bt[n][k] (16 rows x 32 halfs)
// per-lane data (16 consecutive K for one column n) is contiguous -> ds_load_b128 x2
__device__ __forceinline__ v16h load_B16t(const _Float16* s) {
  int lane = threadIdx.x & 31;
  int n = lane & 15;
  int koff = (lane >> 4) << 4;
  v8h lo = *(const v8h*)(s + n * 32 + koff);
  v8h hi = *(const v8h*)(s + n * 32 + koff + 8);
  return __builtin_shufflevector(lo, hi, 0, 1, 2, 3, 4, 5, 6, 7,
                                 8, 9, 10, 11, 12, 13, 14, 15);
}

// ---------- generic WMMA GEMM: C[M,Nc] = act(A[M,K] @ B[K,Nc] + bias) ----------
// 256 threads = 8 waves, one 16x16 C tile per wave, private LDS slice per wave
// (no block barriers: wave-internal DS ordering + s_wait_dscnt). M%16==0, Nc%16==0, K%32==0.
__global__ void gemm_wmma_f16(const float* __restrict__ A, const float* __restrict__ Bw,
                              const float* __restrict__ bias, float* __restrict__ C,
                              int M, int Nc, int K, int act) {
  __shared__ _Float16 sm[8 * (16 * 32 + 16 * 32)];
  const int wave = threadIdx.x >> 5;
  const int lane = threadIdx.x & 31;
  _Float16* As = sm + wave * (16 * 32 + 16 * 32);
  _Float16* Bt = As + 16 * 32;
  const int ntiles = Nc >> 4;
  const int total  = (M >> 4) * ntiles;
  const int t = blockIdx.x * 8 + wave;
  if (t >= total) return;
  const int m0 = (t / ntiles) << 4;
  const int n0 = (t % ntiles) << 4;
  v8f c;
#pragma unroll
  for (int i = 0; i < 8; ++i) c[i] = 0.0f;
  for (int k0 = 0; k0 < K; k0 += 32) {
    // stage A tile 16x32: 128 float4 chunks, 4 per lane (b128 global, b64 LDS stores)
#pragma unroll
    for (int it = 0; it < 4; ++it) {
      int cch = lane + 32 * it;          // chunk id
      int r = cch >> 3, c4 = (cch & 7) << 2;
      v4f av = *(const v4f*)&A[(size_t)(m0 + r) * K + k0 + c4];
      *(v4h*)&As[r * 32 + c4] = cvt4(av);
    }
    // stage B tile 32x16 transposed: 128 float4 chunks (along n), 4 per lane
#pragma unroll
    for (int it = 0; it < 4; ++it) {
      int cch = lane + 32 * it;
      int r = cch >> 2, c4 = (cch & 3) << 2;
      v4f bv = *(const v4f*)&Bw[(size_t)(k0 + r) * Nc + n0 + c4];
#pragma unroll
      for (int j = 0; j < 4; ++j) Bt[(c4 + j) * 32 + r] = (_Float16)bv[j];
    }
    asm volatile("s_wait_dscnt 0x0" ::: "memory");
    v16h a = load_A16(As, 32);
    v16h b = load_B16t(Bt);
    c = __builtin_amdgcn_wmma_f32_16x16x32_f16(false, a, false, b, (short)0, c, false, false);
    asm volatile("s_wait_dscnt 0x0" ::: "memory");
  }
  const int n  = lane & 15;
  const int mb = (lane >> 4) << 3;
  float bv = bias ? bias[n0 + n] : 0.0f;
#pragma unroll
  for (int r = 0; r < 8; ++r) {
    float v = c[r] + bv;
    if (act == 1) v = v > 0.f ? v : 0.f;
    else if (act == 2) v = gelu_f(v);
    C[(size_t)(m0 + mb + r) * Nc + n0 + n] = v;
  }
}

// ---------- fills ----------
__global__ void fill_f32(float* p, float v, int n) {
  int t = blockIdx.x * 256 + threadIdx.x;
  if (t < n) p[t] = v;
}
__global__ void fill_u32(unsigned* p, unsigned v, int n) {
  int t = blockIdx.x * 256 + threadIdx.x;
  if (t < n) p[t] = v;
}

// ---------- embeddings ----------
__global__ void embed_nodes(const int* __restrict__ tok, const float* __restrict__ emb,
                            float* __restrict__ h) {
  int t = blockIdx.x * 256 + threadIdx.x;
  if (t >= N_NODES * D_DIM) return;
  int n = t >> 6, k = t & 63;
  float v = emb[tok[n] * D_DIM + k];
  h[t] = v > 0.f ? v : 0.f;
}
__global__ void embed_edges(const int* __restrict__ tok, const float* __restrict__ emb,
                            float* __restrict__ e) {
  int t = blockIdx.x * 256 + threadIdx.x;
  if (t >= E_EDGES * D_DIM) return;
  int ee = t >> 6, k = t & 63;
  e[t] = emb[tok[ee] * D_DIM + k];
}

// ---------- EGAT edge stages ----------
__global__ void egat_edge_fout(const float* __restrict__ hWni, const float* __restrict__ eW,
                               const float* __restrict__ hWnj, const float* __restrict__ bias,
                               const float* __restrict__ attn, const int* __restrict__ src,
                               const int* __restrict__ dst, float* __restrict__ fout,
                               float* __restrict__ logits, unsigned* __restrict__ mseg) {
  int e = blockIdx.x * 256 + threadIdx.x;
  if (e >= E_EDGES) return;
  int s = src[e], d = dst[e];
  float lg = 0.f;
#pragma unroll 4
  for (int k = 0; k < 64; ++k) {
    float v = hWni[s * 64 + k] + eW[(size_t)e * 64 + k] + hWnj[d * 64 + k] + bias[k];
    v = v > 0.f ? v : 0.01f * v;
    fout[(size_t)e * 64 + k] = v;
    lg += v * attn[k];
  }
  logits[e] = lg;
  atomicMax(&mseg[d], f2ord(lg));
}
__global__ void egat_edge_exp(const float* __restrict__ logits, const int* __restrict__ dst,
                              const unsigned* __restrict__ mseg, float* __restrict__ ex,
                              float* __restrict__ sseg) {
  int e = blockIdx.x * 256 + threadIdx.x;
  if (e >= E_EDGES) return;
  int d = dst[e];
  float x = expf(logits[e] - ord2f(mseg[d]));
  ex[e] = x;
  atomicAdd(&sseg[d], x);
}
__global__ void egat_edge_scatter(const float* __restrict__ ex, const float* __restrict__ sseg,
                                  const float* __restrict__ hWsrc, const int* __restrict__ src,
                                  const int* __restrict__ dst, float* __restrict__ hnew) {
  int t = blockIdx.x * 256 + threadIdx.x;
  if (t >= E_EDGES * D_DIM) return;
  int e = t >> 6, k = t & 63;
  int s = src[e], d = dst[e];
  float a = ex[e] / (sseg[d] + 1e-9f);
  atomicAdd(&hnew[d * 64 + k], a * hWsrc[s * 64 + k]);
}
__global__ void egat_node_finish(const float* __restrict__ hnew, float* __restrict__ h,
                                 float* __restrict__ hs, int l) {
  int t = blockIdx.x * 256 + threadIdx.x;
  if (t >= N_NODES * D_DIM) return;
  int n = t >> 6, k = t & 63;
  float v = hnew[t];
  v = v > 0.f ? v : 0.f;
  h[t] = v;
  hs[(size_t)n * ADIM_ + l * 64 + k] = v;
}

// ---------- attention prep: per (node, head) key scale = 1/((|qk|+1e-8)*sqrt(DH)) ----------
__global__ void key_scales(const float* __restrict__ qk, float* __restrict__ rscale) {
  int t = blockIdx.x * 256 + threadIdx.x;
  if (t >= N_NODES * HEADS_) return;
  int n = t >> 2, hd = t & 3;
  float s = 0.f;
#pragma unroll 8
  for (int k = 0; k < DH_; ++k) {
    float q = qk[(size_t)n * ADIM_ + hd * DH_ + k];
    s += q * q;
  }
  rscale[t] = 1.0f / ((sqrtf(s) + 1e-8f) * 11.3137085f); // sqrt(128)
}

// ---------- flash-style causal shared-QK attention (one wave per head x 16-query tile) ----------
__global__ void flash_attn(const float* __restrict__ qk, const float* __restrict__ vv,
                           const float* __restrict__ rscale, float* __restrict__ ao) {
  __shared__ _Float16 sm[8 * (16 * 128 + 16 * 32)];
  const int wave = threadIdx.x >> 5;
  const int lane = threadIdx.x & 31;
  _Float16* Qs = sm + wave * (16 * 128 + 16 * 32);
  _Float16* Ps = Qs + 16 * 128;
  const int gw = blockIdx.x * 8 + wave;   // 1024 waves total
  const int qt = gw >> 2;
  const int hd = gw & 3;
  const int q0 = qt * 16;
  const int hv = lane >> 4;
  const int n  = lane & 15;

  // stage Q tile (16x128) to LDS as f16: 512 float4 chunks, 16 per lane
#pragma unroll
  for (int it = 0; it < 16; ++it) {
    int cch = lane + 32 * it;
    int r = cch >> 5, c4 = (cch & 31) << 2;
    v4f x = *(const v4f*)&qk[(size_t)(q0 + r) * ADIM_ + hd * DH_ + c4];
    *(v4h*)&Qs[r * 128 + c4] = cvt4(x);
  }
  asm volatile("s_wait_dscnt 0x0" ::: "memory");

  v16h qa[4];
#pragma unroll
  for (int dc = 0; dc < 4; ++dc) qa[dc] = load_A16(Qs + dc * 32, 128);

  float mrow[8], lrow[8];
  v8f acc[8];
#pragma unroll
  for (int i = 0; i < 8; ++i) {
    mrow[i] = -3.0e38f;
    lrow[i] = 0.f;
#pragma unroll
    for (int j = 0; j < 8; ++j) acc[i][j] = 0.f;
  }

  const int ktmax = (q0 + 15) >> 5;
  for (int kt = 0; kt <= ktmax; ++kt) {
    const int k0 = kt * 32;
    v8f sC[2];
    // S = (Q . K^T) with k-norm and 1/sqrt(DH) folded into the B operand
#pragma unroll
    for (int t2 = 0; t2 < 2; ++t2) {
      v8f c;
#pragma unroll
      for (int j = 0; j < 8; ++j) c[j] = 0.f;
      const int key = k0 + t2 * 16 + n;
      const float ks = rscale[key * 4 + hd];
      const size_t kbase = (size_t)key * ADIM_ + hd * DH_;
      const int koff = hv << 4;
#pragma unroll
      for (int dc = 0; dc < 4; ++dc) {
        v16h b;
#pragma unroll
        for (int q4 = 0; q4 < 4; ++q4) {
          v4f x = *(const v4f*)&qk[kbase + dc * 32 + koff + q4 * 4];
          b[q4 * 4 + 0] = (_Float16)(x[0] * ks);
          b[q4 * 4 + 1] = (_Float16)(x[1] * ks);
          b[q4 * 4 + 2] = (_Float16)(x[2] * ks);
          b[q4 * 4 + 3] = (_Float16)(x[3] * ks);
        }
        c = __builtin_amdgcn_wmma_f32_16x16x32_f16(false, qa[dc], false, b, (short)0, c, false, false);
      }
      sC[t2] = c;
    }
    // mask + online softmax (rows r -> global row q0 + r + 8*hv; cols -> k0 + t2*16 + n)
#pragma unroll
    for (int r = 0; r < 8; ++r) {
      const int row = q0 + r + 8 * hv;
      float best = -3.0e38f;
#pragma unroll
      for (int t2 = 0; t2 < 2; ++t2) {
        const int col = k0 + t2 * 16 + n;
        float sv = sC[t2][r];
        sv = (col > row) ? -1e9f : ((col == row) ? -5e4f : sv);
        sC[t2][r] = sv;
        best = fmaxf(best, sv);
      }
      for (int mm = 1; mm < 16; mm <<= 1) best = fmaxf(best, __shfl_xor(best, mm, 32));
      const float mnew = fmaxf(mrow[r], best);
      const float corr = expf(mrow[r] - mnew);
      float psum = 0.f;
#pragma unroll
      for (int t2 = 0; t2 < 2; ++t2) {
        float p = expf(sC[t2][r] - mnew);
        sC[t2][r] = p;
        psum += p;
      }
      for (int mm = 1; mm < 16; mm <<= 1) psum += __shfl_xor(psum, mm, 32);
      lrow[r] = lrow[r] * corr + psum;
      mrow[r] = mnew;
#pragma unroll
      for (int dc8 = 0; dc8 < 8; ++dc8) acc[dc8][r] *= corr;
    }
    // P (C-layout) -> LDS -> A-layout f16
#pragma unroll
    for (int r = 0; r < 8; ++r) {
      const int rl = r + 8 * hv;
      Ps[rl * 32 + n]      = (_Float16)sC[0][r];
      Ps[rl * 32 + 16 + n] = (_Float16)sC[1][r];
    }
    asm volatile("s_wait_dscnt 0x0" ::: "memory");
    v16h pa = load_A16(Ps, 32);
    const int koff = hv << 4;
#pragma unroll
    for (int dc8 = 0; dc8 < 8; ++dc8) {
      v16h b;
#pragma unroll
      for (int i = 0; i < 16; ++i)
        b[i] = (_Float16)vv[(size_t)(k0 + koff + i) * ADIM_ + hd * DH_ + dc8 * 16 + n];
      acc[dc8] = __builtin_amdgcn_wmma_f32_16x16x32_f16(false, pa, false, b, (short)0, acc[dc8], false, false);
    }
    asm volatile("s_wait_dscnt 0x0" ::: "memory");
  }
  // write out: ao[row, hd*128 + dc8*16 + n]
#pragma unroll
  for (int dc8 = 0; dc8 < 8; ++dc8) {
#pragma unroll
    for (int r = 0; r < 8; ++r) {
      const int row = q0 + r + 8 * hv;
      ao[(size_t)row * ADIM_ + hd * DH_ + dc8 * 16 + n] = acc[dc8][r] / lrow[r];
    }
  }
}

// ---------- SortPooling ----------
__global__ void sortpool_sort(const float* __restrict__ h3, float* __restrict__ xs) {
  int node = blockIdx.x * 256 + threadIdx.x;
  if (node >= N_NODES) return;
  float vals[64];
#pragma unroll
  for (int k = 0; k < 64; ++k) vals[k] = h3[(size_t)node * 64 + k];
  for (int i = 1; i < 64; ++i) {
    float key = vals[i];
    int j = i - 1;
    while (j >= 0 && vals[j] > key) { vals[j + 1] = vals[j]; --j; }
    vals[j + 1] = key;
  }
#pragma unroll
  for (int k = 0; k < 64; ++k) xs[(size_t)node * 64 + k] = vals[k];
}
__global__ void sortpool_topk(const float* __restrict__ xs, float* __restrict__ pooled) {
  int g = blockIdx.x * 32 + threadIdx.x;
  if (g >= B_GRAPH) return;
  bool used[NPG_];
  for (int i = 0; i < NPG_; ++i) used[i] = false;
  for (int kk = 0; kk < KPOOL; ++kk) {
    float best = -3.0e38f;
    int bi = 0;
    for (int nl = 0; nl < NPG_; ++nl) {
      if (used[nl]) continue;
      float v = xs[(size_t)(g * NPG_ + nl) * 64 + 63];
      if (v > best) { best = v; bi = nl; }
    }
    used[bi] = true;
    for (int k = 0; k < 64; ++k)
      pooled[(size_t)g * (KPOOL * 64) + kk * 64 + k] = xs[(size_t)(g * NPG_ + bi) * 64 + k];
  }
}

// ---------- fg GAT ----------
__global__ void gat_scores(const float* __restrict__ f, const float* __restrict__ al,
                           const float* __restrict__ ar, float* __restrict__ el,
                           float* __restrict__ er) {
  int g = blockIdx.x * 32 + threadIdx.x;
  if (g >= B_GRAPH) return;
  float a = 0.f, b = 0.f;
#pragma unroll
  for (int k = 0; k < 64; ++k) {
    a += f[g * 64 + k] * al[k];
    b += f[g * 64 + k] * ar[k];
  }
  el[g] = a;
  er[g] = b;
}
__global__ void fg_logit(const float* __restrict__ el, const float* __restrict__ er,
                         const int* __restrict__ fs, const int* __restrict__ fd,
                         float* __restrict__ lg, unsigned* __restrict__ m2) {
  int e = blockIdx.x * 256 + threadIdx.x;
  if (e >= EFG_) return;
  float x = el[fs[e]] + er[fd[e]];
  x = x > 0.f ? x : 0.2f * x;
  lg[e] = x;
  atomicMax(&m2[fd[e]], f2ord(x));
}
__global__ void fg_exp(const float* __restrict__ lg, const int* __restrict__ fd,
                       const unsigned* __restrict__ m2, float* __restrict__ ex2,
                       float* __restrict__ s2) {
  int e = blockIdx.x * 256 + threadIdx.x;
  if (e >= EFG_) return;
  float x = expf(lg[e] - ord2f(m2[fd[e]]));
  ex2[e] = x;
  atomicAdd(&s2[fd[e]], x);
}
__global__ void fg_scatter(const float* __restrict__ ex2, const float* __restrict__ s2,
                           const float* __restrict__ f, const int* __restrict__ fs,
                           const int* __restrict__ fd, float* __restrict__ hgacc) {
  int t = blockIdx.x * 256 + threadIdx.x;
  if (t >= EFG_ * 64) return;
  int e = t >> 6, k = t & 63;
  float a = ex2[e] / (s2[fd[e]] + 1e-9f);
  atomicAdd(&hgacc[fd[e] * 64 + k], a * f[fs[e] * 64 + k]);
}
__global__ void hg_finish(const float* __restrict__ hgacc, const float* __restrict__ gb,
                          float* __restrict__ hg) {
  int t = blockIdx.x * 256 + threadIdx.x;
  if (t >= B_GRAPH * 64) return;
  float v = hgacc[t] + gb[t & 63];
  hg[t] = v > 0.f ? v : 0.f;
}
__global__ void final_cls(const float* __restrict__ x, const float* __restrict__ Wc,
                          const float* __restrict__ bc, float* __restrict__ out) {
  int t = blockIdx.x * 64 + threadIdx.x;
  if (t >= B_GRAPH * 2) return;
  int g = t >> 1, c = t & 1;
  float s = bc[c];
#pragma unroll
  for (int k = 0; k < 64; ++k) s += x[g * 64 + k] * Wc[k * 2 + c];
  out[t] = s;
}

// ---------- host orchestration ----------
extern "C" void kernel_launch(void* const* d_in, const int* in_sizes, int n_in,
                              void* d_out, int out_size, void* d_ws, size_t ws_size,
                              hipStream_t stream) {
  const int*   h_tok    = (const int*)d_in[0];
  const int*   e_tok    = (const int*)d_in[1];
  const int*   src      = (const int*)d_in[2];
  const int*   dst      = (const int*)d_in[3];
  const int*   fg_src   = (const int*)d_in[4];
  const int*   fg_dst   = (const int*)d_in[5];
  const float* tok_emb  = (const float*)d_in[6];
  const float* etok_emb = (const float*)d_in[7];
  const float* Wsrc     = (const float*)d_in[8];
  const float* Wni      = (const float*)d_in[9];
  const float* Wfij     = (const float*)d_in[10];
  const float* Wnj      = (const float*)d_in[11];
  const float* eattn    = (const float*)d_in[12];
  const float* ebias    = (const float*)d_in[13];
  const float* Wqk      = (const float*)d_in[14];
  const float* Wv       = (const float*)d_in[15];
  const float* Wout     = (const float*)d_in[16];
  const float* bout     = (const float*)d_in[17];
  const float* Wi1      = (const float*)d_in[18];
  const float* bi1      = (const float*)d_in[19];
  const float* Wi2      = (const float*)d_in[20];
  const float* bi2      = (const float*)d_in[21];
  const float* Wf1      = (const float*)d_in[22];
  const float* bf1      = (const float*)d_in[23];
  const float* gat_W    = (const float*)d_in[24];
  const float* gat_al   = (const float*)d_in[25];
  const float* gat_ar   = (const float*)d_in[26];
  const float* gat_b    = (const float*)d_in[27];
  const float* Wl       = (const float*)d_in[28];
  const float* bl       = (const float*)d_in[29];
  const float* Wl1      = (const float*)d_in[30];
  const float* bl1      = (const float*)d_in[31];
  const float* Wc       = (const float*)d_in[32];
  const float* bc       = (const float*)d_in[33];
  float* out = (float*)d_out;

  char* ws = (char*)d_ws;
  size_t off = 0;
  auto alloc = [&](size_t elems) -> char* {
    char* p = ws + off;
    off = (off + elems * 4 + 255) & ~(size_t)255;
    return p;
  };
  float*    h      = (float*)alloc(N_NODES * 64);
  float*    hWni   = (float*)alloc(N_NODES * 64);
  float*    hWnj   = (float*)alloc(N_NODES * 64);
  float*    hWsrc  = (float*)alloc(N_NODES * 64);
  float*    hnew   = (float*)alloc(N_NODES * 64);
  float*    e1     = (float*)alloc((size_t)E_EDGES * 64);
  float*    e2     = (float*)alloc((size_t)E_EDGES * 64);
  float*    eW     = (float*)alloc((size_t)E_EDGES * 64);
  float*    logits = (float*)alloc(E_EDGES);
  float*    exbuf  = (float*)alloc(E_EDGES);
  unsigned* mseg   = (unsigned*)alloc(N_NODES);
  float*    sseg   = (float*)alloc(N_NODES);
  float*    hs     = (float*)alloc((size_t)N_NODES * ADIM_);
  float*    qkb    = (float*)alloc((size_t)N_NODES * ADIM_);
  float*    vvb    = (float*)alloc((size_t)N_NODES * ADIM_);
  float*    rsc    = (float*)alloc(N_NODES * HEADS_);
  float*    ao     = (float*)alloc((size_t)N_NODES * ADIM_);
  float*    h2     = (float*)alloc((size_t)N_NODES * ADIM_);
  float*    t1     = (float*)alloc(N_NODES * 64);
  float*    t2b    = (float*)alloc((size_t)N_NODES * ADIM_);
  float*    h3     = (float*)alloc(N_NODES * 64);
  float*    xs     = (float*)alloc(N_NODES * 64);
  float*    pooled = (float*)alloc(B_GRAPH * KPOOL * 64);
  float*    fgf    = (float*)alloc(B_GRAPH * 64);
  float*    el     = (float*)alloc(B_GRAPH);
  float*    er     = (float*)alloc(B_GRAPH);
  float*    lgb    = (float*)alloc(EFG_);
  float*    ex2    = (float*)alloc(EFG_);
  unsigned* m2     = (unsigned*)alloc(B_GRAPH);
  float*    s2     = (float*)alloc(B_GRAPH);
  float*    hgacc  = (float*)alloc(B_GRAPH * 64);
  float*    hg     = (float*)alloc(B_GRAPH * 64);
  float*    tA     = (float*)alloc(B_GRAPH * 64);
  float*    tB     = (float*)alloc(B_GRAPH * 64);
  (void)ws_size; (void)n_in; (void)in_sizes; (void)out_size;

  auto gemm = [&](const float* A, const float* W, const float* bias, float* C,
                  int M, int Nc, int K, int act) {
    int tiles = (M / 16) * (Nc / 16);
    gemm_wmma_f16<<<dim3((tiles + 7) / 8), dim3(256), 0, stream>>>(A, W, bias, C, M, Nc, K, act);
  };

  // embeddings
  embed_nodes<<<dim3(N_NODES * 64 / 256), dim3(256), 0, stream>>>(h_tok, tok_emb, h);
  embed_edges<<<dim3(E_EDGES * 64 / 256), dim3(256), 0, stream>>>(e_tok, etok_emb, e1);

  float* ecur = e1;
  float* enext = e2;
  for (int l = 0; l < L_LAYERS; ++l) {
    const float* Wni_l  = Wni  + (size_t)l * 64 * 64;
    const float* Wnj_l  = Wnj  + (size_t)l * 64 * 64;
    const float* Wsrc_l = Wsrc + (size_t)l * 64 * 64;
    const float* Wfij_l = Wfij + (size_t)l * 64 * 64;
    gemm(h, Wni_l,  nullptr, hWni,  N_NODES, 64, 64, 0);
    gemm(h, Wnj_l,  nullptr, hWnj,  N_NODES, 64, 64, 0);
    gemm(h, Wsrc_l, nullptr, hWsrc, N_NODES, 64, 64, 0);
    gemm(ecur, Wfij_l, nullptr, eW, E_EDGES, 64, 64, 0);
    fill_u32<<<dim3(N_NODES / 256), dim3(256), 0, stream>>>(mseg, 0u, N_NODES);
    fill_f32<<<dim3(N_NODES / 256), dim3(256), 0, stream>>>(sseg, 0.f, N_NODES);
    fill_f32<<<dim3(N_NODES * 64 / 256), dim3(256), 0, stream>>>(hnew, 0.f, N_NODES * 64);
    egat_edge_fout<<<dim3(E_EDGES / 256), dim3(256), 0, stream>>>(
        hWni, eW, hWnj, ebias + l * 64, eattn + l * 64, src, dst, enext, logits, mseg);
    egat_edge_exp<<<dim3(E_EDGES / 256), dim3(256), 0, stream>>>(logits, dst, mseg, exbuf, sseg);
    egat_edge_scatter<<<dim3(E_EDGES * 64 / 256), dim3(256), 0, stream>>>(
        exbuf, sseg, hWsrc, src, dst, hnew);
    egat_node_finish<<<dim3(N_NODES * 64 / 256), dim3(256), 0, stream>>>(hnew, h, hs, l);
    float* tmp = ecur; ecur = enext; enext = tmp;
  }

  // attention
  gemm(hs, Wqk, nullptr, qkb, N_NODES, ADIM_, ADIM_, 0);
  gemm(hs, Wv,  nullptr, vvb, N_NODES, ADIM_, ADIM_, 0);
  key_scales<<<dim3(N_NODES * HEADS_ / 256), dim3(256), 0, stream>>>(qkb, rsc);
  flash_attn<<<dim3(HEADS_ * (N_NODES / 16) / 8), dim3(256), 0, stream>>>(qkb, vvb, rsc, ao);

  // post-attention MLPs
  gemm(ao, Wout, bout, h2, N_NODES, ADIM_, ADIM_, 0);
  gemm(h2, Wi1, bi1, t1,  N_NODES, 64,    ADIM_, 2);
  gemm(t1, Wi2, bi2, t2b, N_NODES, ADIM_, 64,    2);
  gemm(t2b, Wf1, bf1, h3, N_NODES, 64,    ADIM_, 1);

  // sort-pooling
  sortpool_sort<<<dim3(N_NODES / 256), dim3(256), 0, stream>>>(h3, xs);
  sortpool_topk<<<dim3(1), dim3(32), 0, stream>>>(xs, pooled);

  // fg GAT
  gemm(pooled, gat_W, nullptr, fgf, B_GRAPH, 64, KPOOL * 64, 0);
  gat_scores<<<dim3(1), dim3(32), 0, stream>>>(fgf, gat_al, gat_ar, el, er);
  fill_u32<<<dim3(1), dim3(256), 0, stream>>>(m2, 0u, B_GRAPH);
  fill_f32<<<dim3(1), dim3(256), 0, stream>>>(s2, 0.f, B_GRAPH);
  fill_f32<<<dim3(B_GRAPH * 64 / 256), dim3(256), 0, stream>>>(hgacc, 0.f, B_GRAPH * 64);
  fg_logit<<<dim3(1), dim3(256), 0, stream>>>(el, er, fg_src, fg_dst, lgb, m2);
  fg_exp<<<dim3(1), dim3(256), 0, stream>>>(lgb, fg_dst, m2, ex2, s2);
  fg_scatter<<<dim3(EFG_ * 64 / 256), dim3(256), 0, stream>>>(ex2, s2, fgf, fg_src, fg_dst, hgacc);
  hg_finish<<<dim3(B_GRAPH * 64 / 256), dim3(256), 0, stream>>>(hgacc, gat_b, hg);

  // classifier head
  gemm(hg, Wl,  bl,  tA, B_GRAPH, 64, 64, 1);
  gemm(tA, Wl1, bl1, tB, B_GRAPH, 64, 64, 1);
  final_cls<<<dim3(1), dim3(64), 0, stream>>>(tB, Wc, bc, out);
}
